// MultiHeadAttention_75419625717781
// MI455X (gfx1250) — matrix operand order repaired and checked by reading, not compiled, not run on hardware
//
#include <hip/hip_runtime.h>

typedef __attribute__((ext_vector_type(16))) __bf16 v16bf;
typedef __attribute__((ext_vector_type(8)))  float  v8f;
typedef __attribute__((ext_vector_type(4)))  unsigned int u32x4;
typedef __attribute__((ext_vector_type(8)))  int i32x8;
typedef __attribute__((ext_vector_type(4)))  int i32x4;
typedef unsigned short u16;
typedef unsigned int   u32;
typedef unsigned long long u64;

#define N_HEAD   16
#define HEAD_DIM 64
#define SEQ_T    2048
#define BATCH    4
#define EMBD     1024
// Faithful to source: 1/sqrt((3*1024)/16) = 1/sqrt(192)
#define QKV_SCALE 0.07216878364870323f

// ---------- helpers ----------

__device__ __forceinline__ u16 f2bf(float f) {
  u32 u = __builtin_bit_cast(u32, f);
  u += 0x7FFFu + ((u >> 16) & 1u);   // round-to-nearest-even truncation
  return (u16)(u >> 16);
}

union FragCast { v16bf v; uint4 q[2]; };

// A-operand fragment (16x32 bf16, tile row-major [M][K]):
// lane L: row = L%16 ; VGPR0-3 hold K dwords [h*4,h*4+4), VGPR4-7 hold [8+h*4,...)
__device__ __forceinline__ v16bf load_frag_a(const u16* tile, int stride_us) {
  const int lane = threadIdx.x & 31;
  const int hl = lane >> 4, r = lane & 15;
  const u32* p = (const u32*)(tile + (size_t)r * stride_us);
  FragCast f;
  f.q[0] = *(const uint4*)(p + hl * 4);
  f.q[1] = *(const uint4*)(p + hl * 4 + 8);
  return f.v;
}

// B-operand fragment (K=32 x N=16 bf16, tile staged as [N][K]):
// lane L: col n = L%16 ; VGPR0-3 hold K dwords [h*4,h*4+4), VGPR4-7 hold [8+h*4,...)
__device__ __forceinline__ v16bf load_frag_b(const u16* tile, int stride_us) {
  const int lane = threadIdx.x & 31;
  const int hl = lane >> 4, n = lane & 15;
  const u32* p = (const u32*)(tile + (size_t)n * stride_us);
  FragCast f;
  f.q[0] = *(const uint4*)(p + hl * 4);
  f.q[1] = *(const uint4*)(p + hl * 4 + 8);
  return f.v;
}

#define WMMA_BF16(A, B, C) \
  __builtin_amdgcn_wmma_f32_16x16x32_bf16(false, (A), false, (B), (short)0, (C), false, false)

// raw LDS byte offset of a __shared__ object (generic -> addrspace(3) -> int)
__device__ __forceinline__ u32 lds_off(const void* p) {
  return (u32)(size_t)(__attribute__((address_space(3))) const void*)p;
}

// ---------- Tensor Data Mover: 2D tile load (bf16 elements) ----------
// dims/tiles/strides in elements (data_size = 2 bytes). D# packing per ISA 8.3/8.4.
__device__ __forceinline__ void tdm_load_2d(u32 lds_addr, const void* gptr,
                                            u32 dim0, u32 dim1,
                                            u32 tile0, u32 tile1, u32 stride0) {
  const u64 ga = (u64)(size_t)gptr;
  u32x4 g0;
  g0[0] = 1u;                                         // count=1, user descriptor
  g0[1] = lds_addr;                                   // lds_addr [63:32]
  g0[2] = (u32)ga;                                    // global_addr[31:0]
  g0[3] = (u32)((ga >> 32) & 0x01FFFFFFu) | (2u << 30); // addr[56:32], type=2
  i32x8 g1;
  g1[0] = (int)(1u << 16);                            // data_size=1 (2 bytes)
  g1[1] = (int)((dim0 & 0xFFFFu) << 16);              // tensor_dim0 lo16 at [63:48]
  g1[2] = (int)((dim0 >> 16) | ((dim1 & 0xFFFFu) << 16));
  g1[3] = (int)((dim1 >> 16) | (tile0 << 16));        // tile_dim0 at [127:112]
  g1[4] = (int)tile1;                                 // tile_dim1, tile_dim2=0
  g1[5] = (int)stride0;                               // tensor_dim0_stride[31:0]
  g1[6] = 0;
  g1[7] = 0;
  const i32x4 z4 = {0, 0, 0, 0};
#if __clang_major__ >= 23
  const i32x8 z8 = {0, 0, 0, 0, 0, 0, 0, 0};
  __builtin_amdgcn_tensor_load_to_lds(g0, g1, z4, z4, z8, 0);
#else
  __builtin_amdgcn_tensor_load_to_lds(g0, g1, z4, z4, 0);
#endif
}

// ---------- generic tiled bf16 WMMA GEMM: C = A @ W + bias ----------
// 256 threads = 8 waves (2 M x 4 N), tile 128x128x32, reg-level pipelining.
// A_IS_BF16: 0 -> A fp32 (converted while staging), 1 -> A already bf16
// EPI_QKV:   0 -> fp32 out + bias, 1 -> split q/k/v (q scaled, v transposed) bf16
template<int A_IS_BF16, int EPI_QKV>
__global__ __launch_bounds__(256) void gemm_bf16_wmma(
    const void* __restrict__ Ap, const float* __restrict__ W,
    const float* __restrict__ bias, float* __restrict__ outF,
    u16* __restrict__ qws, u16* __restrict__ kws, u16* __restrict__ vws,
    int M, int N, int K, float qscale)
{
  __shared__ alignas(16) u16 As[128 * 32];   // [M][K] bf16
  __shared__ alignas(16) u16 Bs[128 * 32];   // [N][K] bf16 (W transposed in staging)
  const int tid  = threadIdx.x;
  const int wid  = tid >> 5;
  const int lane = tid & 31;
  const int hl = lane >> 4, ln = lane & 15;
  const int wm = wid & 1, wn = wid >> 1;     // wave grid 2 x 4
  const int m0 = blockIdx.y * 128;
  const int n0 = blockIdx.x * 128;

  v8f acc[8];
#pragma unroll
  for (int i = 0; i < 8; ++i)
#pragma unroll
    for (int r = 0; r < 8; ++r) acc[i][r] = 0.0f;

  float4 ra[4];   // fp32-A pipeline regs
  uint4  rabf[2]; // bf16-A pipeline regs
  float4 rb[4];   // W pipeline regs

  auto loadA = [&](int k0) {
    if (A_IS_BF16) {
      const u16* A = (const u16*)Ap;
      const int r = tid >> 1, cd = (tid & 1) * 8;
      const u32* src = (const u32*)(A + (size_t)(m0 + r) * K + k0) + cd;
      rabf[0] = *(const uint4*)(src);
      rabf[1] = *(const uint4*)(src + 4);
    } else {
      const float* A = (const float*)Ap;
#pragma unroll
      for (int i = 0; i < 4; ++i) {
        const int f = i * 256 + tid;       // 1024 float4s (128x32 fp32)
        const int r = f >> 3, c4 = f & 7;
        ra[i] = *(const float4*)(A + (size_t)(m0 + r) * K + k0 + c4 * 4);
      }
    }
  };
  auto loadB = [&](int k0) {
#pragma unroll
    for (int i = 0; i < 4; ++i) {
      const int f = i * 256 + tid;         // 1024 float4s (32x128 fp32)
      const int kk = f >> 5, n4 = f & 31;
      rb[i] = *(const float4*)(W + (size_t)(k0 + kk) * N + n0 + n4 * 4);
    }
  };
  auto storeA = [&]() {
    if (A_IS_BF16) {
      const int r = tid >> 1, cd = (tid & 1) * 8;
      u32* dst = (u32*)As + r * 16 + cd;
      *(uint4*)(dst)     = rabf[0];
      *(uint4*)(dst + 4) = rabf[1];
    } else {
      u32* Asw = (u32*)As;
#pragma unroll
      for (int i = 0; i < 4; ++i) {
        const int f = i * 256 + tid;
        const int r = f >> 3, c4 = f & 7;
        Asw[r * 16 + c4 * 2]     = (u32)f2bf(ra[i].x) | ((u32)f2bf(ra[i].y) << 16);
        Asw[r * 16 + c4 * 2 + 1] = (u32)f2bf(ra[i].z) | ((u32)f2bf(ra[i].w) << 16);
      }
    }
  };
  auto storeB = [&]() {
#pragma unroll
    for (int i = 0; i < 4; ++i) {
      const int f = i * 256 + tid;
      const int kk = f >> 5, n4 = f & 31;
      Bs[(n4 * 4 + 0) * 32 + kk] = f2bf(rb[i].x);
      Bs[(n4 * 4 + 1) * 32 + kk] = f2bf(rb[i].y);
      Bs[(n4 * 4 + 2) * 32 + kk] = f2bf(rb[i].z);
      Bs[(n4 * 4 + 3) * 32 + kk] = f2bf(rb[i].w);
    }
  };

  loadA(0); loadB(0);
  for (int k0 = 0; k0 < K; k0 += 32) {
    storeA(); storeB();
    __syncthreads();
    if (k0 + 32 < K) { loadA(k0 + 32); loadB(k0 + 32); }  // hide global latency
    v16bf bfr[2];
#pragma unroll
    for (int nt = 0; nt < 2; ++nt)
      bfr[nt] = load_frag_b(Bs + (wn * 32 + nt * 16) * 32, 32);
#pragma unroll
    for (int j = 0; j < 4; ++j) {
      const v16bf a = load_frag_a(As + (wm * 64 + j * 16) * 32, 32);
#pragma unroll
      for (int nt = 0; nt < 2; ++nt)
        acc[j * 2 + nt] = WMMA_BF16(a, bfr[nt], acc[j * 2 + nt]);
    }
    __syncthreads();
  }

  // ---- epilogue ----
#pragma unroll
  for (int j = 0; j < 4; ++j)
#pragma unroll
    for (int nt = 0; nt < 2; ++nt)
#pragma unroll
      for (int r = 0; r < 8; ++r) {
        const int m = m0 + wm * 64 + j * 16 + hl * 8 + r;
        const int n = n0 + wn * 32 + nt * 16 + ln;
        const float v = acc[j * 2 + nt][r] + bias[n];
        if (EPI_QKV == 0) {
          outF[(size_t)m * N + n] = v;
        } else {
          const int sec = n >> 10, c = n & 1023;
          const int h = c >> 6, d = c & 63;
          const int b = m >> 11, t = m & (SEQ_T - 1);
          if (sec == 0) {        // q: [B,H,T,D], pre-scaled
            qws[(((size_t)(b * N_HEAD + h)) * SEQ_T + t) * HEAD_DIM + d] = f2bf(v * qscale);
          } else if (sec == 1) { // k: [B,H,T,D]
            kws[(((size_t)(b * N_HEAD + h)) * SEQ_T + t) * HEAD_DIM + d] = f2bf(v);
          } else {               // v: [B,H,D,T]  (pre-transposed for TDM staging)
            vws[(((size_t)(b * N_HEAD + h)) * HEAD_DIM + d) * SEQ_T + t] = f2bf(v);
          }
        }
      }
}

// ---------- flash attention: one block = one (b,h) x 64 Q rows ----------
// KV blocks of 64, double-buffered LDS filled by the Tensor Data Mover.
__global__ __launch_bounds__(128) void flash_attn_wmma(
    const u16* __restrict__ qws, const u16* __restrict__ kws,
    const u16* __restrict__ vws, u16* __restrict__ yws)
{
  __shared__ alignas(16) u16 Kbuf[2][64 * 64];   // [s][d]
  __shared__ alignas(16) u16 Vbuf[2][64 * 64];   // [d][s]
  __shared__ alignas(16) u16 Pt[4][16 * 64];     // per-wave P tile [m][s]
  const int tid  = threadIdx.x;
  const int wid  = tid >> 5;
  const int lane = tid & 31;
  const int hl = lane >> 4, ln = lane & 15;
  const int bh = blockIdx.x;                     // b*16 + h
  const int t0 = blockIdx.y * 64;
  const u16* kblk = kws + (size_t)bh * SEQ_T * HEAD_DIM;   // [s][d]
  const u16* vblk = vws + (size_t)bh * HEAD_DIM * SEQ_T;   // [d][t]

  // Q fragments in registers (A operand), 16 rows/wave, K-dim 64 = 2 steps
  const u16* qrow = qws + (size_t)bh * SEQ_T * HEAD_DIM + (size_t)(t0 + wid * 16) * HEAD_DIM;
  const v16bf qa0 = load_frag_a(qrow, HEAD_DIM);
  const v16bf qa1 = load_frag_a(qrow + 32, HEAD_DIM);

  // all-ones B fragment: rowsum(P) = P @ ones (folds softmax denom into WMMA)
  FragCast onesf;
  onesf.q[0] = make_uint4(0x3F803F80u, 0x3F803F80u, 0x3F803F80u, 0x3F803F80u);
  onesf.q[1] = onesf.q[0];
  const v16bf ones = onesf.v;

  float mrow[8];
  v8f o[4], osum;
#pragma unroll
  for (int r = 0; r < 8; ++r) { mrow[r] = -1e30f; osum[r] = 0.0f; }
#pragma unroll
  for (int i = 0; i < 4; ++i)
#pragma unroll
    for (int r = 0; r < 8; ++r) o[i][r] = 0.0f;

  const int NBLK = SEQ_T / 64;
  if (wid == 0) {  // TDM: wave 0 drives the DMA engine; EXEC-independent
    tdm_load_2d(lds_off(&Kbuf[0][0]), kblk, HEAD_DIM, SEQ_T, 64, 64, HEAD_DIM);
    tdm_load_2d(lds_off(&Vbuf[0][0]), vblk, SEQ_T, HEAD_DIM, 64, 64, SEQ_T);
  }

  for (int i = 0; i < NBLK; ++i) {
    const int cur = i & 1, nxt = cur ^ 1;
    if (wid == 0) {
      if (i + 1 < NBLK) {  // prefetch next KV block into the other buffer
        tdm_load_2d(lds_off(&Kbuf[nxt][0]), kblk + (size_t)(i + 1) * 64 * HEAD_DIM,
                    HEAD_DIM, SEQ_T, 64, 64, HEAD_DIM);
        tdm_load_2d(lds_off(&Vbuf[nxt][0]), vblk + (size_t)(i + 1) * 64,
                    SEQ_T, HEAD_DIM, 64, 64, SEQ_T);
        __builtin_amdgcn_s_wait_tensorcnt(2);  // block i done (in-order TDM)
      } else {
        __builtin_amdgcn_s_wait_tensorcnt(0);
      }
    }
    __syncthreads();

    const u16* Kt = &Kbuf[cur][0];
    const u16* Vt = &Vbuf[cur][0];

    // S = Q @ K^T : 16 x 64 per wave (4 col-tiles x 2 k-steps)
    v8f S[4];
#pragma unroll
    for (int nt = 0; nt < 4; ++nt) {
#pragma unroll
      for (int r = 0; r < 8; ++r) S[nt][r] = 0.0f;
      S[nt] = WMMA_BF16(qa0, load_frag_b(Kt + nt * 16 * 64, 64), S[nt]);
      S[nt] = WMMA_BF16(qa1, load_frag_b(Kt + nt * 16 * 64 + 32, 64), S[nt]);
    }

    // online softmax: butterfly only for the row max; sums go through WMMA
    float alpha[8];
#pragma unroll
    for (int r = 0; r < 8; ++r) {
      float v = fmaxf(fmaxf(S[0][r], S[1][r]), fmaxf(S[2][r], S[3][r]));
#pragma unroll
      for (int msk = 1; msk < 16; msk <<= 1) v = fmaxf(v, __shfl_xor(v, msk, 32));
      const float mn = fmaxf(mrow[r], v);
      alpha[r] = __expf(mrow[r] - mn);
      mrow[r] = mn;
#pragma unroll
      for (int nt = 0; nt < 4; ++nt) {
        const float p = __expf(S[nt][r] - mn);
        Pt[wid][(hl * 8 + r) * 64 + nt * 16 + ln] = f2bf(p);  // C-layout -> [m][s]
      }
    }
#pragma unroll
    for (int nt = 0; nt < 4; ++nt)
#pragma unroll
      for (int r = 0; r < 8; ++r) o[nt][r] *= alpha[r];
#pragma unroll
    for (int r = 0; r < 8; ++r) osum[r] *= alpha[r];

    // O += P @ V ; osum += P @ ones
    const v16bf pa0 = load_frag_a(&Pt[wid][0], 64);
    const v16bf pa1 = load_frag_a(&Pt[wid][0] + 32, 64);
#pragma unroll
    for (int nt = 0; nt < 4; ++nt) {
      o[nt] = WMMA_BF16(pa0, load_frag_b(Vt + nt * 16 * 64, 64), o[nt]);
      o[nt] = WMMA_BF16(pa1, load_frag_b(Vt + nt * 16 * 64 + 32, 64), o[nt]);
    }
    osum = WMMA_BF16(pa0, ones, osum);
    osum = WMMA_BF16(pa1, ones, osum);
    __syncthreads();
  }

  // epilogue: y[b, t, h*64 + d] bf16 (merged-head layout for proj GEMM)
  const int b = bh >> 4, h = bh & 15;
#pragma unroll
  for (int nt = 0; nt < 4; ++nt)
#pragma unroll
    for (int r = 0; r < 8; ++r) {
      const int t = t0 + wid * 16 + hl * 8 + r;
      const int c = h * 64 + nt * 16 + ln;
      yws[((size_t)b * SEQ_T + t) * EMBD + c] = f2bf(o[nt][r] / osum[r]);
    }
}

// ---------- launch ----------
extern "C" void kernel_launch(void* const* d_in, const int* in_sizes, int n_in,
                              void* d_out, int out_size, void* d_ws, size_t ws_size,
                              hipStream_t stream) {
  (void)in_sizes; (void)n_in; (void)out_size; (void)ws_size;
  const float* x      = (const float*)d_in[0];
  const float* W_attn = (const float*)d_in[1];
  const float* b_attn = (const float*)d_in[2];
  const float* W_proj = (const float*)d_in[3];
  const float* b_proj = (const float*)d_in[4];
  float* out = (float*)d_out;

  const size_t qkv_elems = (size_t)BATCH * N_HEAD * SEQ_T * HEAD_DIM; // 8.4M bf16 each
  u16* qws = (u16*)d_ws;                  // 16 MB  [B,H,T,D]
  u16* kws = qws + qkv_elems;             // 16 MB  [B,H,T,D]
  u16* vws = kws + qkv_elems;             // 16 MB  [B,H,D,T] (transposed)
  u16* yws = vws + qkv_elems;             // 16 MB  [B*T, C]

  // 1) QKV projection: [8192,1024] @ [1024,3072] + b, split/scale/transpose epilogue
  gemm_bf16_wmma<0, 1><<<dim3(24, 64), 256, 0, stream>>>(
      (const void*)x, W_attn, b_attn, nullptr, qws, kws, vws,
      BATCH * SEQ_T, 3 * EMBD, EMBD, QKV_SCALE);

  // 2) flash attention per (b,h) x 64-row Q block, TDM-staged KV
  flash_attn_wmma<<<dim3(BATCH * N_HEAD, SEQ_T / 64), 128, 0, stream>>>(
      qws, kws, vws, yws);

  // 3) output projection: [8192,1024] @ [1024,1024] + b -> fp32 out
  gemm_bf16_wmma<1, 0><<<dim3(8, 64), 256, 0, stream>>>(
      (const void*)yws, W_proj, b_proj, out, nullptr, nullptr, nullptr,
      BATCH * SEQ_T, EMBD, EMBD, 1.0f);
}
